// HFF_78692390797356
// MI455X (gfx1250) — compile-verified, matrix operand order checked by reading, not run on hardware
//
#include <hip/hip_runtime.h>

// ---------------------------------------------------------------------------
// Problem constants (from reference): B=2, C=256, H=W=64
// ---------------------------------------------------------------------------
#define PB   2
#define CC   256
#define HH   64
#define WW   64
#define PP   4096          // H*W
#define NSET 128
#define WCBN 6144          // 2*C*cg*k
#define NSUB (WCBN / 16)   // 384 column subtiles in the swizzled codebook

typedef __attribute__((ext_vector_type(16))) _Float16 v16h;
typedef __attribute__((ext_vector_type(8)))  _Float16 v8h;   // 16B, align 16
typedef __attribute__((ext_vector_type(4)))  _Float16 v4h;   // 8B
typedef __attribute__((ext_vector_type(8)))  float    v8f;
typedef __attribute__((ext_vector_type(4)))  float    v4f;   // 16B

union H16 { v16h v; v8h q[2]; _Float16 h[16]; };
union F8  { v8f  v; float f[8]; };

// ---------------------------------------------------------------------------
// Mean over H,W per (b,c)
// ---------------------------------------------------------------------------
__global__ void mean_kernel(const float* __restrict__ x, float* __restrict__ mean) {
    __shared__ float red[256];
    int bc = blockIdx.x;                       // 0 .. B*C-1
    const float* p = x + (size_t)bc * PP;
    float s = 0.f;
    for (int i = threadIdx.x; i < PP; i += 256) s += p[i];
    red[threadIdx.x] = s;
    __syncthreads();
    for (int off = 128; off > 0; off >>= 1) {
        if ((int)threadIdx.x < off) red[threadIdx.x] += red[threadIdx.x + off];
        __syncthreads();
    }
    if (threadIdx.x == 0) mean[bc] = red[0] * (1.0f / (float)PP);
}

// sca = sca_w @ mean + sca_b   (tiny: 512 outputs)
__global__ void sca_kernel(const float* __restrict__ mean,
                           const float* __restrict__ w,
                           const float* __restrict__ bias,
                           float* __restrict__ sca) {
    int b  = blockIdx.x;
    int co = threadIdx.x;
    float acc = bias[co];
    const float* wr = w + (size_t)co * CC;
    const float* mr = mean + (size_t)b * CC;
    for (int ci = 0; ci < CC; ++ci) acc += wr[ci] * mr[ci];
    sca[b * CC + co] = acc;
}

// ---------------------------------------------------------------------------
// Generic 1x1-conv GEMM: Out[b][co][p] = sum_ci Wg(co,ci) * X[b][ci][p] (+bias)
// Wg element (co,ci) at Wg[co*wsr + ci*wsc]  (wsr=Cin,wsc=1 normal; 1,Cout transp)
// f32 -> f16 staged in LDS (float4 global loads), v_wmma_f32_16x16x32_f16.
// A tile [m][k]; B tile TRANSPOSED [n][k] so fragments are two 16-byte runs
// per lane -> ds_load_b128.
// Block tile: 32 (co) x 64 (p); 8 waves, one 16x16 subtile each.
// grid = (P/64, Cout/32, B)
// ---------------------------------------------------------------------------
__global__ __launch_bounds__(256) void gemm_wmma(
    const float* __restrict__ X, const float* __restrict__ Wg,
    const float* __restrict__ bias, float* __restrict__ Out,
    int Cin, int Cout, int wsr, int wsc)
{
    __shared__ __align__(16) _Float16 lA [32][40];   // [m][k], 80B row stride
    __shared__ __align__(16) _Float16 lBt[64][40];   // [n][k] transposed

    const int b   = blockIdx.z;
    const int bm0 = blockIdx.y << 5;
    const int bn0 = blockIdx.x << 6;
    const int tid = threadIdx.x;
    const int wave = tid >> 5, lane = tid & 31;
    const int msub = (wave >> 2) << 4;       // 0 or 16
    const int nsub = (wave & 3) << 4;        // 0,16,32,48
    const int ml = lane & 15;                // row/col within subtile
    const int hk = (lane >> 4) << 3;         // K-half select per WMMA layout

    F8 acc;
#pragma unroll
    for (int r = 0; r < 8; ++r) acc.f[r] = 0.f;

    const float* Xb = X + (size_t)b * Cin * PP;

    for (int k0 = 0; k0 < Cin; k0 += 32) {
        // ---- A: 32x32, f32 -> f16 ----
        if (wsc == 1) {                              // row-major: float4 + b64 store
            int m = tid >> 3, kk4 = (tid & 7) << 2;
            v4f a4 = *(const v4f*)&Wg[(size_t)(bm0 + m) * wsr + k0 + kk4];
            v4h h4;
#pragma unroll
            for (int j = 0; j < 4; ++j) h4[j] = (_Float16)a4[j];
            *(v4h*)&lA[m][kk4] = h4;
        } else {                                     // transposed weights (b_cb)
#pragma unroll
            for (int t = 0; t < 4; ++t) {
                int i = tid + (t << 8);
                int m = i >> 5, kk = i & 31;
                lA[m][kk] = (_Float16)Wg[(size_t)(bm0 + m) * wsr + (size_t)(k0 + kk) * wsc];
            }
        }
        // ---- B: 32x64, float4 global load, transposed LDS store ----
#pragma unroll
        for (int t = 0; t < 2; ++t) {
            int i = tid + (t << 8);                  // 0..511
            int kk = i >> 4, n4 = (i & 15) << 2;
            v4f b4 = *(const v4f*)&Xb[(size_t)(k0 + kk) * PP + bn0 + n4];
#pragma unroll
            for (int j = 0; j < 4; ++j) lBt[n4 + j][kk] = (_Float16)b4[j];
        }
        if (k0 + 32 < Cin)                           // next-tile prefetch
            __builtin_prefetch(&Xb[(size_t)(k0 + 32) * PP + bn0 + (tid & 63)], 0, 1);
        __syncthreads();

        H16 af, bf;
        af.q[0] = *(const v8h*)&lA [msub + ml][hk];
        af.q[1] = *(const v8h*)&lA [msub + ml][hk + 16];
        bf.q[0] = *(const v8h*)&lBt[nsub + ml][hk];
        bf.q[1] = *(const v8h*)&lBt[nsub + ml][hk + 16];
        acc.v = __builtin_amdgcn_wmma_f32_16x16x32_f16(
            false, af.v, false, bf.v, (short)0, acc.v, false, false);
        __syncthreads();
    }

    const int mh = (lane >> 4) << 3;         // C/D layout: VGPR r -> M=r (+8 hi lanes)
    size_t obase = (size_t)b * Cout * PP + bn0 + nsub + ml;
    if (bias) {
#pragma unroll
        for (int r = 0; r < 8; ++r) {
            int m = bm0 + msub + mh + r;
            Out[obase + (size_t)m * PP] = acc.f[r] + bias[m];
        }
    } else {
#pragma unroll
        for (int r = 0; r < 8; ++r) {
            int m = bm0 + msub + mh + r;
            Out[obase + (size_t)m * PP] = acc.f[r];
        }
    }
}

// ---------------------------------------------------------------------------
// Depthwise conv (square kernel, dilation), no bias. One thread per output.
// ---------------------------------------------------------------------------
__global__ void dwconv_kernel(const float* __restrict__ in,
                              const float* __restrict__ wgt,
                              float* __restrict__ out,
                              int ks, int pad, int dil) {
    int idx = blockIdx.x * 256 + threadIdx.x;
    if (idx >= PB * CC * PP) return;
    int w = idx & 63, h = (idx >> 6) & 63, c = (idx >> 12) & 255, b = idx >> 20;
    const float* wp = wgt + (size_t)c * ks * ks;
    const float* ip = in + (((size_t)b * CC + c) << 12);
    float acc = 0.f;
    for (int kh = 0; kh < ks; ++kh) {
        int hh = h + kh * dil - pad;
        if ((unsigned)hh >= (unsigned)HH) continue;
        for (int kw = 0; kw < ks; ++kw) {
            int ww2 = w + kw * dil - pad;
            if ((unsigned)ww2 >= (unsigned)WW) continue;
            acc += wp[kh * ks + kw] * ip[(hh << 6) + ww2];
        }
    }
    out[idx] = acc;
}

// Grouped conv c2a: 32 groups, 8 in-ch per group, 1 out-ch per group, 3x3 pad1
__global__ void grouped_c2a_kernel(const float* __restrict__ x,
                                   const float* __restrict__ wgt,
                                   const float* __restrict__ bias,
                                   float* __restrict__ out) {
    int idx = blockIdx.x * 256 + threadIdx.x;
    if (idx >= PB * 32 * PP) return;
    int w = idx & 63, h = (idx >> 6) & 63, co = (idx >> 12) & 31, b = idx >> 17;
    float acc = bias[co];
    const float* wp = wgt + (size_t)co * 72;             // 8*3*3
    const float* ip = x + (((size_t)b * CC + co * 8) << 12);
    for (int ci = 0; ci < 8; ++ci) {
        for (int kh = 0; kh < 3; ++kh) {
            int hh = h + kh - 1;
            if ((unsigned)hh >= (unsigned)HH) continue;
            for (int kw = 0; kw < 3; ++kw) {
                int ww2 = w + kw - 1;
                if ((unsigned)ww2 >= (unsigned)WW) continue;
                acc += wp[ci * 9 + kh * 3 + kw] * ip[((size_t)ci << 12) + (hh << 6) + ww2];
            }
        }
    }
    out[idx] = acc;
}

// SimpleGate: sg[b,c,p] = t[b,c,p] * t[b,c+16,p]
__global__ void gate_kernel(const float* __restrict__ t, float* __restrict__ sg) {
    int idx = blockIdx.x * 256 + threadIdx.x;
    if (idx >= PB * 16 * PP) return;
    int p = idx & 4095, c = (idx >> 12) & 15, b = idx >> 16;
    const float* tb = t + (((size_t)b * 32) << 12);
    sg[idx] = tb[((size_t)c << 12) + p] * tb[((size_t)(c + 16) << 12) + p];
}

// att += attgamma[n] * (c2b_b[n] + sum_ci sg[b,ci,p]*c2b_w[n,ci])   (Cin=16)
__global__ void c2b_fuse_kernel(const float* __restrict__ sg,
                                const float* __restrict__ w,
                                const float* __restrict__ bias,
                                const float* __restrict__ gamma,
                                float* att) {
    int idx = blockIdx.x * 256 + threadIdx.x;
    if (idx >= PB * NSET * PP) return;
    int p = idx & 4095, n = (idx >> 12) & 127, b = idx >> 19;
    float acc = bias[n];
    const float* sb = sg + (((size_t)b * 16) << 12) + p;
    const float* wr = w + (size_t)n * 16;
    for (int ci = 0; ci < 16; ++ci) acc += wr[ci] * sb[(size_t)ci << 12];
    att[idx] += gamma[n] * acc;
}

// ---------------------------------------------------------------------------
// Pre-swizzle w_cb (fp32 [128][6144]) into exact WMMA B-fragment order (f16):
// for subtile st (16 cols), k-block kb (32 k), lane l: 16 contiguous halves =
// the lane's fragment. ikba then loads B with two coalesced global_load_b128.
// ---------------------------------------------------------------------------
__global__ void swizzle_wcb_kernel(const float* __restrict__ w_cb,
                                   _Float16* __restrict__ out) {
    int idx = blockIdx.x * 256 + threadIdx.x;        // (st*4+kb)*32+lane
    if (idx >= NSUB * 4 * 32) return;
    int lane = idx & 31;
    int kb   = (idx >> 5) & 3;
    int st   = idx >> 7;
    int hk   = (lane >> 4) << 3;
    int nl   = lane & 15;
    _Float16* o = out + ((size_t)idx << 4);
#pragma unroll
    for (int e = 0; e < 16; ++e) {
        int k = (kb << 5) + hk + (e < 8 ? e : e + 8);
        o[e] = (_Float16)w_cb[(size_t)k * WCBN + (st << 4) + nl];
    }
}

// ---------------------------------------------------------------------------
// Fused IKBA pass: for 16 pixels, WMMA-compute attk tile [16 x 768] into LDS
// (chunk of 64 output channels x 12 taps), then immediately contract with the
// unfolded taps. attk (201 MB fp32) is never materialized in HBM.
//   vertical  (horizontal=0): taps along H from src=uf;   out = xh
//   horizontal(horizontal=1): taps along W from src=xh;   out = (acc+bias)*ga1+uf
// grid = (P/16, B), 256 threads (8 waves: 6 N-subtiles each, K=128 in 4 steps)
// ---------------------------------------------------------------------------
__global__ __launch_bounds__(256) void ikba_pass_kernel(
    const float* __restrict__ att,       // [B][128][P]
    const _Float16* __restrict__ wcb2,   // swizzled codebook fragments
    const float* __restrict__ src,       // tap source
    const float* biasbuf,                // [B][C][P]  (horizontal only; may alias out)
    const float* __restrict__ ufb,       // uf for final blend (horizontal only)
    const float* __restrict__ ga1,       // [C]
    float* out,
    int horizontal)
{
    __shared__ __align__(16) _Float16 lA[16][136];   // a-tile: 16 px x 128 s (272B rows)
    __shared__ float lD[16][768];                    // attk chunk: 16 px x (64 co * 12)

    const int b   = blockIdx.y;
    const int p0  = blockIdx.x << 4;
    const int tid = threadIdx.x;
    const int wave = tid >> 5, lane = tid & 31;
    const int ml = lane & 15;
    const int hk = (lane >> 4) << 3;
    const int mh = (lane >> 4) << 3;
    const int stbase = (horizontal ? (WCBN / 32) : 0);   // 192 subtiles per half

    // load A tile (att transposed to [pixel][s]), float4 over contiguous pixels
#pragma unroll
    for (int t = 0; t < 2; ++t) {
        int i = tid + (t << 8);          // 0..511
        int s = i >> 2, pl4 = (i & 3) << 2;
        v4f a4 = *(const v4f*)&att[((size_t)b * NSET + s) * PP + p0 + pl4];
#pragma unroll
        for (int j = 0; j < 4; ++j) lA[pl4 + j][s] = (_Float16)a4[j];
    }
    __syncthreads();

    for (int chunk = 0; chunk < 4; ++chunk) {
        // ---- GEMM: 48 subtiles of 16 columns, 6 per wave ----
#pragma unroll
        for (int t = 0; t < 6; ++t) {
            int nst = wave * 6 + t;                       // 0..47
            int st  = stbase + chunk * 48 + nst;          // absolute subtile
            F8 acc;
#pragma unroll
            for (int r = 0; r < 8; ++r) acc.f[r] = 0.f;
#pragma unroll
            for (int kb = 0; kb < 4; ++kb) {
                H16 af, bf;
                af.q[0] = *(const v8h*)&lA[ml][(kb << 5) + hk];
                af.q[1] = *(const v8h*)&lA[ml][(kb << 5) + hk + 16];
                const v8h* wp = (const v8h*)(wcb2 +
                    ((((size_t)((st << 2) + kb) << 5) + lane) << 4));
                bf.q[0] = wp[0];
                bf.q[1] = wp[1];
                acc.v = __builtin_amdgcn_wmma_f32_16x16x32_f16(
                    false, af.v, false, bf.v, (short)0, acc.v, false, false);
            }
#pragma unroll
            for (int r = 0; r < 8; ++r)
                lD[mh + r][(nst << 4) + ml] = acc.f[r];
        }
        __syncthreads();

        // ---- apply: 16 pixels x 64 channels in this chunk ----
#pragma unroll
        for (int t = 0; t < 4; ++t) {
            int i  = tid + (t << 8);         // 0..1023
            int pl = i & 15, col = i >> 4;   // col 0..63
            int co = (chunk << 6) + col;
            int g4 = co >> 2;                // group index
            int p  = p0 + pl;
            int h  = p >> 6, w = p & 63;
            const float* sb = src + (((size_t)b * CC + (g4 << 2)) << 12);
            float acc = 0.f;
#pragma unroll
            for (int cl = 0; cl < 4; ++cl) {
#pragma unroll
                for (int tap = 0; tap < 3; ++tap) {
                    float v;
                    if (!horizontal) {
                        int hh = h + tap - 1;
                        v = ((unsigned)hh < (unsigned)HH)
                              ? sb[((size_t)cl << 12) + (hh << 6) + w] : 0.f;
                    } else {
                        int ww2 = w + tap - 1;
                        v = ((unsigned)ww2 < (unsigned)WW)
                              ? sb[((size_t)cl << 12) + (h << 6) + ww2] : 0.f;
                    }
                    acc += lD[pl][col * 12 + cl * 3 + tap] * v;
                }
            }
            size_t oidx = ((size_t)b * CC + co) * PP + p;
            if (!horizontal) {
                out[oidx] = acc;
            } else {
                out[oidx] = (acc + biasbuf[oidx]) * ga1[co] + ufb[oidx];
            }
        }
        __syncthreads();
    }
}

// prod = x1 * x2 * sca[b,c] * lka   (float4 vectorized; 4 px share (b,c))
__global__ void mult_kernel(const float* __restrict__ x1, const float* __restrict__ x2,
                            const float* __restrict__ lka, const float* __restrict__ sca,
                            float* __restrict__ out) {
    int idx = blockIdx.x * 256 + threadIdx.x;        // over TOT/4
    if (idx >= (PB * CC * PP) / 4) return;
    int e = idx << 2;
    int c = (e >> 12) & 255, b = e >> 20;
    v4f r = ((const v4f*)x1)[idx] * ((const v4f*)x2)[idx]
          * ((const v4f*)lka)[idx] * sca[b * CC + c];
    ((v4f*)out)[idx] = r;
}

// ---------------------------------------------------------------------------
extern "C" void kernel_launch(void* const* d_in, const int* in_sizes, int n_in,
                              void* d_out, int out_size, void* d_ws, size_t ws_size,
                              hipStream_t stream) {
    const float* x        = (const float*)d_in[0];
    const float* dw1_w    = (const float*)d_in[1];
    const float* dw2_w    = (const float*)d_in[2];
    const float* proj_w   = (const float*)d_in[3];
    const float* lka0_w   = (const float*)d_in[4];
    const float* lkas_w   = (const float*)d_in[5];
    const float* lka1_w   = (const float*)d_in[6];
    const float* lka1_b   = (const float*)d_in[7];
    const float* sca_w    = (const float*)d_in[8];
    const float* sca_b    = (const float*)d_in[9];
    const float* c1a_w    = (const float*)d_in[10];
    const float* c1b_w    = (const float*)d_in[11];
    const float* c2a_w    = (const float*)d_in[12];
    const float* c2a_b    = (const float*)d_in[13];
    const float* c2b_w    = (const float*)d_in[14];
    const float* c2b_b    = (const float*)d_in[15];
    const float* c211_w   = (const float*)d_in[16];
    const float* c211_b   = (const float*)d_in[17];
    const float* w_cb     = (const float*)d_in[18];
    const float* b_cb     = (const float*)d_in[19];
    const float* attgamma = (const float*)d_in[20];
    const float* ga1      = (const float*)d_in[21];

    float* ws = (float*)d_ws;
    const size_t S = (size_t)PB * CC * PP;           // 2,097,152 floats
    float* W0  = ws;                                  // v1 -> lka7 -> u0 -> bias -> x2
    float* W1  = ws + 1 * S;                          // x1
    float* W2  = ws + 2 * S;                          // lka5 -> lka
    float* W3  = ws + 3 * S;                          // uf  -> prod
    float* W5  = ws + 4 * S;                          // xh
    float* ATT = ws + 5 * S;                          // [B][128][P]
    float* T   = ATT + (size_t)PB * NSET * PP;        // [B][32][P]
    float* SG  = T   + (size_t)PB * 32 * PP;          // [B][16][P]
    float* MEAN= SG  + (size_t)PB * 16 * PP;          // 512
    float* SCA = MEAN + 512;                          // 512
    _Float16* WCB2 = (_Float16*)(SCA + 512);          // swizzled codebook (1.5 MB)

    const int TOT  = PB * CC * PP;
    const int GRID = (TOT + 255) / 256;

    // mean + sca
    mean_kernel<<<PB * CC, 256, 0, stream>>>(x, MEAN);
    sca_kernel<<<PB, 256, 0, stream>>>(MEAN, sca_w, sca_b, SCA);

    // x1 branch: v1 = dw1(x); x1 = dw3x3(v1)
    gemm_wmma<<<dim3(PP / 64, CC / 32, PB), 256, 0, stream>>>(x, dw1_w, nullptr, W0, CC, CC, CC, 1);
    dwconv_kernel<<<GRID, 256, 0, stream>>>(W0, dw2_w, W1, 3, 1, 1);

    // lka branch: dw5x5 -> dw7x7 dil3 -> 1x1(+b)
    dwconv_kernel<<<GRID, 256, 0, stream>>>(x,  lka0_w, W2, 5, 2, 1);
    dwconv_kernel<<<GRID, 256, 0, stream>>>(W2, lkas_w, W0, 7, 9, 3);
    gemm_wmma<<<dim3(PP / 64, CC / 32, PB), 256, 0, stream>>>(W0, lka1_w, lka1_b, W2, CC, CC, CC, 1);

    // uf = dw3x3(c1a(x))
    gemm_wmma<<<dim3(PP / 64, CC / 32, PB), 256, 0, stream>>>(x, c1a_w, nullptr, W0, CC, CC, CC, 1);
    dwconv_kernel<<<GRID, 256, 0, stream>>>(W0, c1b_w, W3, 3, 1, 1);

    // attention: att = c2b(SimpleGate(c2a(x)))*gamma + c211(x)
    grouped_c2a_kernel<<<(PB * 32 * PP + 255) / 256, 256, 0, stream>>>(x, c2a_w, c2a_b, T);
    gate_kernel<<<(PB * 16 * PP + 255) / 256, 256, 0, stream>>>(T, SG);
    gemm_wmma<<<dim3(PP / 64, NSET / 32, PB), 256, 0, stream>>>(x, c211_w, c211_b, ATT, CC, NSET, CC, 1);
    c2b_fuse_kernel<<<(PB * NSET * PP + 255) / 256, 256, 0, stream>>>(SG, c2b_w, c2b_b, attgamma, ATT);

    // ikba: codebook swizzled to WMMA fragment order; bias = b_cb^T @ a
    swizzle_wcb_kernel<<<(NSUB * 4 * 32 + 255) / 256, 256, 0, stream>>>(w_cb, WCB2);
    gemm_wmma<<<dim3(PP / 64, CC / 32, PB), 256, 0, stream>>>(ATT, b_cb, nullptr, W0, NSET, CC, 1, CC);

    // fused attk-GEMM + dynamic-conv passes (attk never hits HBM)
    ikba_pass_kernel<<<dim3(PP / 16, PB), 256, 0, stream>>>(ATT, WCB2, W3, nullptr, nullptr, nullptr, W5, 0);
    ikba_pass_kernel<<<dim3(PP / 16, PB), 256, 0, stream>>>(ATT, WCB2, W5, W0, W3, ga1, W0, 1);

    // prod = x1 * x2 * sca * lka ; out = proj(prod)
    mult_kernel<<<(TOT / 4 + 255) / 256, 256, 0, stream>>>(W1, W0, W2, SCA, W3);
    gemm_wmma<<<dim3(PP / 64, CC / 32, PB), 256, 0, stream>>>(W3, proj_w, nullptr, (float*)d_out, CC, CC, CC, 1);
}
